// UnifiedFlowFrag_30777735643335
// MI455X (gfx1250) — matrix operand order repaired and checked by reading, not compiled, not run on hardware
//
#include <hip/hip_runtime.h>
#include <hip/hip_bf16.h>

#define N_NODES 10000
#define N_EDGES 160000
#define S_DIM   256
#define V_DIM   32
#define L2_DIM  16
#define TDIM_C  128
#define FEAT    528
#define ESD_C   700
#define ESD_PAD 704
#define WOUT_C  400
#define ADA_DIM 592
#define DEG_INV_C 0.25f
#define ETILE   32     // edges per workgroup (2 WMMA M-subtiles)

typedef __attribute__((ext_vector_type(16))) __bf16 v16bf;
typedef __attribute__((ext_vector_type(8)))  float  v8f;

union ABFrag { v16bf v; uint4 q[2]; };

__device__ __forceinline__ unsigned short f2bf(float f) {
  unsigned u = __float_as_uint(f);
  u += 0x7fffu + ((u >> 16) & 1u);          // round-to-nearest-even
  return (unsigned short)(u >> 16);
}

__device__ __forceinline__ void atomic_add_f32(float* p, float v) {
  __hip_atomic_fetch_add(p, v, __ATOMIC_RELAXED, __HIP_MEMORY_SCOPE_AGENT);
}

// A fragment (16x32 bf16, M x K) from LDS, per ISA 7.12.2 16-bit A layout:
// lanes 0-15: row M=lane, K = kbase+{0..7} and kbase+{16..23}
// lanes 16-31: row M=lane-16, K = kbase+{8..15} and kbase+{24..31}
__device__ __forceinline__ v16bf load_a_frag_lds(const unsigned short* Ash, int ldk, int kbase) {
  int lane = threadIdx.x & 31;
  int m = lane & 15;
  int half8 = (lane >> 4) * 8;
  ABFrag u;
  u.q[0] = *(const uint4*)(Ash + m * ldk + kbase + half8);
  u.q[1] = *(const uint4*)(Ash + m * ldk + kbase + 16 + half8);
  return u.v;
}

// B fragment (32x16 bf16, K x N) from the pre-packed layout (one 32B record per lane per tile)
__device__ __forceinline__ v16bf load_b_frag(const unsigned short* Bp, int numNT, int kt, int nt) {
  int lane = threadIdx.x & 31;
  const uint4* p = (const uint4*)Bp + (size_t)(((kt * numNT + nt) * 32 + lane)) * 2;
  ABFrag u;
  u.q[0] = p[0];
  u.q[1] = p[1];
  return u.v;
}

// -------- weight packing: fp32 row-major KxN  ->  bf16 WMMA B-fragment tiles --------
__global__ void pack_b_kernel(const float* __restrict__ B, int K, int N, int Kpad,
                              unsigned short* __restrict__ out) {
  int t = blockIdx.x * blockDim.x + threadIdx.x;
  int numNT = N >> 4;
  int total = (Kpad >> 5) * numNT * 32;
  if (t >= total) return;
  int lane = t & 31;
  int tile = t >> 5;
  int nt = tile % numNT;
  int kt = tile / numNT;
  int n = nt * 16 + (lane & 15);
  int khalf = (lane >> 4) * 8;
  unsigned short* o = out + (size_t)t * 16;
#pragma unroll
  for (int j = 0; j < 8; ++j) {
    int k0 = kt * 32 + khalf + ((2 * j) & 7) + 16 * (j >> 2);
    int k1 = k0 + 1;
    o[2*j]   = (k0 < K) ? f2bf(B[(size_t)k0 * N + n]) : (unsigned short)0;
    o[2*j+1] = (k1 < K) ? f2bf(B[(size_t)k1 * N + n]) : (unsigned short)0;
  }
}

// -------- generic M-tiled bf16 WMMA GEMM: C[M x N] = (ascale*A[M x K]) @ Bpack + bias --------
// block = 256 (8 waves), one 16-row M tile per block, waves stride over N tiles.
__global__ void gemm_bf16_wmma(const float* __restrict__ A, int lda, int K, int Kpad, float ascale,
                               const unsigned short* __restrict__ Bp, int N,
                               const float* __restrict__ bias,
                               float* __restrict__ C) {
  extern __shared__ __align__(16) unsigned short Ash[];
  int tid = threadIdx.x;
  int rowBase = blockIdx.x * 16;
  for (int i = tid; i < 16 * Kpad; i += 256) {
    int m = i / Kpad;
    int k = i - m * Kpad;
    float v = (k < K) ? A[(size_t)(rowBase + m) * lda + k] * ascale : 0.0f;
    Ash[m * Kpad + k] = f2bf(v);
  }
  __syncthreads();
  int wave = tid >> 5;
  int lane = tid & 31;
  int numNT = N >> 4;
  int numKT = Kpad >> 5;
  for (int nt = wave; nt < numNT; nt += 8) {
    v8f acc = {};
    for (int kt = 0; kt < numKT; ++kt) {
      v16bf a = load_a_frag_lds(Ash, Kpad, kt * 32);
      v16bf b = load_b_frag(Bp, numNT, kt, nt);
      acc = __builtin_amdgcn_wmma_f32_16x16x32_bf16(false, a, false, b, (short)0, acc, false, false);
    }
    int col = nt * 16 + (lane & 15);
    float bv = bias ? bias[col] : 0.0f;
    int mBase = (lane >> 4) * 8;
#pragma unroll
    for (int r = 0; r < 8; ++r)
      C[(size_t)(rowBase + mBase + r) * N + col] = acc[r] + bv;
  }
}

// -------- fused edge kernel: 32 edges/block, build 32x704 bf16 features in LDS,
//          dual-accumulator WMMA (B fragment reused 2x) vs edge_W, messages + atomics --------
__global__ void edge_fused_kernel(
    const float* __restrict__ h, const float* __restrict__ coords,
    const float* __restrict__ erd, const float* __restrict__ t_emb,
    const float* __restrict__ ety, const float* __restrict__ bty,
    const float* __restrict__ bcj, const float* __restrict__ brg,
    const float* __restrict__ bst,
    const float* __restrict__ refW, const float* __restrict__ refb,
    const unsigned short* __restrict__ Bp, const float* __restrict__ eb,
    const float* __restrict__ sv,
    const int* __restrict__ eidx, const int* __restrict__ et,
    const int* __restrict__ ebt, const int* __restrict__ ebc,
    const int* __restrict__ ebr, const int* __restrict__ ebs,
    float* __restrict__ agg_s, float* __restrict__ agg_v1o,
    float* __restrict__ agg_v1e, float* __restrict__ agg_l2) {
  __shared__ __align__(16) unsigned short Ash[ETILE * ESD_PAD];  // 44 KB bf16 A tile
  __shared__ float Wsh[ETILE * WOUT_C];                          // 50 KB edge MLP output
  __shared__ int   Ssrc[ETILE], Sdst[ETILE];
  __shared__ float Sdist[ETILE], SY1[ETILE][3], Sref[ETILE][8];
  __shared__ int   Sidx[ETILE][5];
  int tid = threadIdx.x;
  int e0 = blockIdx.x * ETILE;

  if (tid < ETILE) {
    int g = e0 + tid;
    int s = eidx[g];
    int d = eidx[N_EDGES + g];
    Ssrc[tid] = s; Sdst[tid] = d;
    float dx = coords[d*3+0] - coords[s*3+0];
    float dy = coords[d*3+1] - coords[s*3+1];
    float dz = coords[d*3+2] - coords[s*3+2];
    float dist = sqrtf(dx*dx + dy*dy + dz*dz);
    Sdist[tid] = dist;
    float invd = 1.0f / (dist + 1e-8f);
    SY1[tid][0] = dx*invd; SY1[tid][1] = dy*invd; SY1[tid][2] = dz*invd;
    float rd = erd[g];
    float hr = rd > 0.0f ? 1.0f : 0.0f;
    float del = dist - rd;
    float f0 = fabsf(del), f1 = del, f2 = hr;
#pragma unroll
    for (int j = 0; j < 8; ++j)
      Sref[tid][j] = f0*refW[j] + f1*refW[8+j] + f2*refW[16+j] + refb[j];
    Sidx[tid][0] = et[g];  Sidx[tid][1] = ebt[g]; Sidx[tid][2] = ebc[g];
    Sidx[tid][3] = ebr[g]; Sidx[tid][4] = ebs[g];
  }
  __syncthreads();

  // Phase B: build bf16 edge-scalar tile in LDS (gathers are L2-resident)
  for (int i = tid; i < ETILE * ESD_PAD; i += 256) {
    int e = i / ESD_PAD;
    int f = i - e * ESD_PAD;
    float val;
    if (f < 16) {
      float c = (10.0f / 15.0f) * (float)f;
      float z = (Sdist[e] - c) * 1.6f;   // width = 0.625
      val = __expf(-z * z);
    } else if (f < 32)  val = ety[Sidx[e][0]*16 + (f-16)];
    else if (f < 40)    val = bty[Sidx[e][1]*8  + (f-32)];
    else if (f < 44)    val = bcj[Sidx[e][2]*4  + (f-40)];
    else if (f < 48)    val = brg[Sidx[e][3]*4  + (f-44)];
    else if (f < 52)    val = bst[Sidx[e][4]*4  + (f-48)];
    else if (f < 60)    val = Sref[e][f-52];
    else if (f < 316)   val = h[(size_t)Ssrc[e]*FEAT + (f-60)];
    else if (f < 572)   val = h[(size_t)Sdst[e]*FEAT + (f-316)];
    else if (f < 700)   val = t_emb[(size_t)Sdst[e]*TDIM_C + (f-572)];
    else                val = 0.0f;
    Ash[e * ESD_PAD + f] = f2bf(val);
  }
  __syncthreads();

  // Phase C: 32x704 @ 704x400; each wave owns an N tile and both 16-edge M subtiles,
  // so every B fragment load feeds TWO v_wmma_f32_16x16x32_bf16 issues.
  {
    int wave = tid >> 5, lane = tid & 31;
    const int numNT = WOUT_C / 16;    // 25
    const int numKT = ESD_PAD / 32;   // 22
    for (int nt = wave; nt < numNT; nt += 8) {
      v8f acc0 = {};
      v8f acc1 = {};
      for (int kt = 0; kt < numKT; ++kt) {
        v16bf b  = load_b_frag(Bp, numNT, kt, nt);
        v16bf a0 = load_a_frag_lds(Ash,                  ESD_PAD, kt * 32);
        v16bf a1 = load_a_frag_lds(Ash + 16 * ESD_PAD,   ESD_PAD, kt * 32);
        acc0 = __builtin_amdgcn_wmma_f32_16x16x32_bf16(false, a0, false, b, (short)0, acc0, false, false);
        acc1 = __builtin_amdgcn_wmma_f32_16x16x32_bf16(false, a1, false, b, (short)0, acc1, false, false);
      }
      int col = nt * 16 + (lane & 15);
      float bias = eb[col];
      int mBase = (lane >> 4) * 8;
#pragma unroll
      for (int r = 0; r < 8; ++r) {
        Wsh[(mBase + r) * WOUT_C + col]        = acc0[r] + bias;
        Wsh[(16 + mBase + r) * WOUT_C + col]   = acc1[r] + bias;
      }
    }
  }
  __syncthreads();

  // Phase D: messages + hardware f32 atomic scatter-add
  // 32*(256 + 96 + 96 + 80) = 8192+3072+3072+2560 = 16896 items
  for (int i = tid; i < 16896; i += 256) {
    if (i < 8192) {                       // m_s
      int e = i >> 8, c = i & 255;
      float val = Wsh[e*WOUT_C + c] * h[(size_t)Ssrc[e]*FEAT + c];
      atomic_add_f32(&agg_s[(size_t)Sdst[e]*S_DIM + c], val);
    } else if (i < 11264) {               // m_v1o
      int j = i - 8192;
      int e = j / 96, r = j - e*96;
      int c = r / 3, d = r - c*3;
      int s = Ssrc[e];
      float vsrc = h[(size_t)s*FEAT + 256 + c*3 + d];
      float val = Wsh[e*WOUT_C + 256 + c] * vsrc
                + Wsh[e*WOUT_C + 288 + c] * sv[(size_t)s*V_DIM + c] * SY1[e][d];
      atomic_add_f32(&agg_v1o[(size_t)Sdst[e]*96 + c*3 + d], val);
    } else if (i < 14336) {               // m_v1e (with cross(v1o_src, Y1))
      int j = i - 11264;
      int e = j / 96, r = j - e*96;
      int c = r / 3, d = r - c*3;
      int s = Ssrc[e];
      const float* vv = &h[(size_t)s*FEAT + 256 + c*3];
      int d1 = (d + 1) % 3, d2 = (d + 2) % 3;
      float cx = vv[d1]*SY1[e][d2] - vv[d2]*SY1[e][d1];
      float val = Wsh[e*WOUT_C + 320 + c] * h[(size_t)s*FEAT + 352 + c*3 + d]
                + Wsh[e*WOUT_C + 352 + c] * cx;
      atomic_add_f32(&agg_v1e[(size_t)Sdst[e]*96 + c*3 + d], val);
    } else {                              // m_l2
      int j = i - 14336;
      int e = j / 80, r = j - e*80;
      int c = r / 5, d = r - c*5;
      float val = Wsh[e*WOUT_C + 384 + c] * h[(size_t)Ssrc[e]*FEAT + 448 + c*5 + d];
      atomic_add_f32(&agg_l2[(size_t)Sdst[e]*80 + c*5 + d], val);
    }
  }
}

// -------- node update: silu + AdaLN layernorm + small projections + vnorms --------
__global__ void node_update_kernel(const float* __restrict__ h_in, float* __restrict__ h_out,
    const float* __restrict__ agg_v1o, const float* __restrict__ agg_v1e,
    const float* __restrict__ agg_l2,
    const float* __restrict__ usp, const float* __restrict__ ada,
    const float* __restrict__ pv1, const float* __restrict__ pv2,
    const float* __restrict__ pl2) {
  __shared__ float red[256];
  __shared__ float av1o[96], av1e[96], aq[80];
  __shared__ float vno[96], vne[96], qn[80];
  __shared__ float sqo[32], sqe[32], sqq[16];
  int n = blockIdx.x, t = threadIdx.x;

  float u = usp[(size_t)n*S_DIM + t];
  float sil = u / (1.0f + __expf(-u));
  float sval = h_in[(size_t)n*FEAT + t] + sil;

  red[t] = sval; __syncthreads();
  for (int st = 128; st > 0; st >>= 1) { if (t < st) red[t] += red[t + st]; __syncthreads(); }
  float mu = red[0] * (1.0f / 256.0f); __syncthreads();
  float dv = sval - mu;
  red[t] = dv * dv; __syncthreads();
  for (int st = 128; st > 0; st >>= 1) { if (t < st) red[t] += red[t + st]; __syncthreads(); }
  float var = red[0] * (1.0f / 256.0f);
  float inv = rsqrtf(var + 1e-5f);
  float sc = ada[(size_t)n*ADA_DIM + t];
  float sh = ada[(size_t)n*ADA_DIM + 256 + t];
  h_out[(size_t)n*FEAT + t] = dv * inv * (1.0f + sc) + sh;

  if (t < 96) { av1o[t] = agg_v1o[(size_t)n*96 + t] * DEG_INV_C;
                av1e[t] = agg_v1e[(size_t)n*96 + t] * DEG_INV_C; }
  if (t < 80) { aq[t] = agg_l2[(size_t)n*80 + t] * DEG_INV_C; }
  __syncthreads();

  if (t < 96) {
    int d = t / 3, i = t - d*3;
    float a1 = 0.0f, a2 = 0.0f;
#pragma unroll
    for (int c = 0; c < 32; ++c) { a1 += av1o[c*3 + i] * pv1[c*32 + d];
                                   a2 += av1e[c*3 + i] * pv2[c*32 + d]; }
    vno[t] = h_in[(size_t)n*FEAT + 256 + t] + a1;
    vne[t] = h_in[(size_t)n*FEAT + 352 + t] + a2;
  } else if (t < 176) {
    int idx = t - 96;
    int d = idx / 5, i = idx - d*5;
    float a = 0.0f;
#pragma unroll
    for (int c = 0; c < 16; ++c) a += aq[c*5 + i] * pl2[c*16 + d];
    qn[idx] = h_in[(size_t)n*FEAT + 448 + idx] + a;
  }
  __syncthreads();

  if (t < 32)      { float a=vno[t*3],b=vno[t*3+1],c=vno[t*3+2]; sqo[t]=a*a+b*b+c*c; }
  else if (t < 64) { int d=t-32; float a=vne[d*3],b=vne[d*3+1],c=vne[d*3+2]; sqe[d]=a*a+b*b+c*c; }
  else if (t < 80) { int d=t-64; float s2=0.0f;
#pragma unroll
                     for (int i=0;i<5;++i){ float x=qn[d*5+i]; s2+=x*x; } sqq[d]=s2; }
  __syncthreads();

  if (t < 96) {
    float so = 0.0f, se = 0.0f;
#pragma unroll
    for (int c = 0; c < 32; ++c) { so += sqo[c]; se += sqe[c]; }
    float rmso = sqrtf(so * (1.0f/32.0f) + 1e-5f);
    float rmse = sqrtf(se * (1.0f/32.0f) + 1e-5f);
    int d = t / 3;
    float g1 = ada[(size_t)n*ADA_DIM + 512 + d];
    float g2 = ada[(size_t)n*ADA_DIM + 544 + d];
    h_out[(size_t)n*FEAT + 256 + t] = vno[t] / rmso * (1.0f + g1);
    h_out[(size_t)n*FEAT + 352 + t] = vne[t] / rmse * (1.0f + g2);
  } else if (t < 176) {
    int idx = t - 96;
    float sq = 0.0f;
#pragma unroll
    for (int c = 0; c < 16; ++c) sq += sqq[c];
    float rmsq = sqrtf(sq * (1.0f/16.0f) + 1e-5f);
    int d = idx / 5;
    float gl = ada[(size_t)n*ADA_DIM + 576 + d];
    h_out[(size_t)n*FEAT + 448 + idx] = qn[idx] / rmsq * (1.0f + gl);
  }
}

extern "C" void kernel_launch(void* const* d_in, const int* in_sizes, int n_in,
                              void* d_out, int out_size, void* d_ws, size_t ws_size,
                              hipStream_t stream) {
  const float* h0        = (const float*)d_in[0];
  const float* coords    = (const float*)d_in[1];
  const float* erd       = (const float*)d_in[2];
  const float* temb      = (const float*)d_in[3];
  const float* etype_emb = (const float*)d_in[4];
  const float* btype_emb = (const float*)d_in[5];
  const float* bconj_emb = (const float*)d_in[6];
  const float* bring_emb = (const float*)d_in[7];
  const float* bster_emb = (const float*)d_in[8];
  const float* ref_W     = (const float*)d_in[9];
  const float* ref_b     = (const float*)d_in[10];
  const float* edge_W    = (const float*)d_in[11];
  const float* edge_b    = (const float*)d_in[12];
  const float* sv_W      = (const float*)d_in[13];
  const float* proj_s    = (const float*)d_in[14];
  const float* proj_v1   = (const float*)d_in[15];
  const float* proj_v2   = (const float*)d_in[16];
  const float* proj_l2   = (const float*)d_in[17];
  const float* ada_W     = (const float*)d_in[18];
  const float* ada_b     = (const float*)d_in[19];
  const int* edge_index  = (const int*)d_in[20];
  const int* edge_type   = (const int*)d_in[21];
  const int* ebt         = (const int*)d_in[22];
  const int* ebc         = (const int*)d_in[23];
  const int* ebr         = (const int*)d_in[24];
  const int* ebs         = (const int*)d_in[25];
  float* out = (float*)d_out;

  size_t off = 0;
  auto take = [&](size_t bytes) -> void* {
    void* p = (void*)((char*)d_ws + off);
    off += (bytes + 255) & ~(size_t)255;
    return p;
  };
  float* h1              = (float*)take((size_t)N_NODES * FEAT * 4);
  unsigned short* edgeWp = (unsigned short*)take((size_t)2 * ESD_PAD * WOUT_C * 2);
  unsigned short* projsp = (unsigned short*)take((size_t)2 * 256 * 256 * 2);
  unsigned short* adaWp  = (unsigned short*)take((size_t)2 * 128 * ADA_DIM * 2);
  unsigned short* svWp   = (unsigned short*)take((size_t)2 * 256 * 32 * 2);
  float* sv    = (float*)take((size_t)N_NODES * 32 * 4);
  float* agg_s = (float*)take((size_t)N_NODES * 256 * 4);
  float* agg_vo= (float*)take((size_t)N_NODES * 96 * 4);
  float* agg_ve= (float*)take((size_t)N_NODES * 96 * 4);
  float* agg_q = (float*)take((size_t)N_NODES * 80 * 4);
  float* usp   = (float*)take((size_t)N_NODES * 256 * 4);
  float* adab  = (float*)take((size_t)N_NODES * ADA_DIM * 4);

  // Pack weights into bf16 WMMA B-fragment layout (re-done every call; deterministic, cheap)
  for (int l = 0; l < 2; ++l) {
    int tE = (ESD_PAD/32) * (WOUT_C/16) * 32;      // 17600
    pack_b_kernel<<<(tE + 255)/256, 256, 0, stream>>>(
        edge_W + (size_t)l*ESD_C*WOUT_C, ESD_C, WOUT_C, ESD_PAD,
        edgeWp + (size_t)l*ESD_PAD*WOUT_C);
    int tP = (256/32) * (256/16) * 32;             // 4096
    pack_b_kernel<<<(tP + 255)/256, 256, 0, stream>>>(
        proj_s + (size_t)l*256*256, 256, 256, 256, projsp + (size_t)l*256*256);
    int tA = (128/32) * (ADA_DIM/16) * 32;         // 4736
    pack_b_kernel<<<(tA + 255)/256, 256, 0, stream>>>(
        ada_W + (size_t)l*128*ADA_DIM, 128, ADA_DIM, 128, adaWp + (size_t)l*128*ADA_DIM);
    int tS = (256/32) * (32/16) * 32;              // 512
    pack_b_kernel<<<(tS + 255)/256, 256, 0, stream>>>(
        sv_W + (size_t)l*256*32, 256, 32, 256, svWp + (size_t)l*256*32);
  }

  for (int l = 0; l < 2; ++l) {
    const float* hcur = (l == 0) ? h0 : h1;
    float* hnext = (l == 0) ? h1 : out;

    hipMemsetAsync(agg_s,  0, (size_t)N_NODES * 256 * 4, stream);
    hipMemsetAsync(agg_vo, 0, (size_t)N_NODES * 96 * 4, stream);
    hipMemsetAsync(agg_ve, 0, (size_t)N_NODES * 96 * 4, stream);
    hipMemsetAsync(agg_q,  0, (size_t)N_NODES * 80 * 4, stream);

    // sv = s @ sv_W
    gemm_bf16_wmma<<<N_NODES/16, 256, 16*256*2, stream>>>(
        hcur, FEAT, 256, 256, 1.0f, svWp + (size_t)l*256*32, 32, nullptr, sv);

    // fused edge MLP + message scatter (32 edges per block)
    edge_fused_kernel<<<N_EDGES/ETILE, 256, 0, stream>>>(
        hcur, coords, erd, temb,
        etype_emb + (size_t)l*9*16, btype_emb + (size_t)l*6*8,
        bconj_emb + (size_t)l*3*4,  bring_emb + (size_t)l*3*4,
        bster_emb + (size_t)l*5*4,
        ref_W + (size_t)l*3*8, ref_b + (size_t)l*8,
        edgeWp + (size_t)l*ESD_PAD*WOUT_C, edge_b + (size_t)l*WOUT_C,
        sv, edge_index, edge_type, ebt, ebc, ebr, ebs,
        agg_s, agg_vo, agg_ve, agg_q);

    // usp = (DEG_INV * agg_s) @ proj_s   (silu applied in node_update)
    gemm_bf16_wmma<<<N_NODES/16, 256, 16*256*2, stream>>>(
        agg_s, 256, 256, 256, DEG_INV_C, projsp + (size_t)l*256*256, 256, nullptr, usp);

    // ada = t_emb @ ada_W + ada_b
    gemm_bf16_wmma<<<N_NODES/16, 256, 16*128*2, stream>>>(
        temb, TDIM_C, 128, 128, 1.0f, adaWp + (size_t)l*128*ADA_DIM, ADA_DIM,
        ada_b + (size_t)l*ADA_DIM, adab);

    node_update_kernel<<<N_NODES, 256, 0, stream>>>(
        hcur, hnext, agg_vo, agg_ve, agg_q, usp, adab,
        proj_v1 + (size_t)l*32*32, proj_v2 + (size_t)l*32*32,
        proj_l2 + (size_t)l*16*16);
  }
}
